// CrossAttention_47476568490624
// MI455X (gfx1250) — matrix operand order repaired and checked by reading, not compiled, not run on hardware
//
#include <hip/hip_runtime.h>

// ---------------------------------------------------------------------------
// CDNA5 (gfx1250) cross-attention: bf16 WMMA GEMMs + barrier-free flash attn.
// All matrix math on v_wmma_f32_16x16x32_bf16 (wave32, 16x16 tiles, K=32).
// ---------------------------------------------------------------------------

typedef __bf16 bf16;
typedef __attribute__((ext_vector_type(16))) __bf16 v16bf;
typedef __attribute__((ext_vector_type(8)))  float  v8f;
typedef __attribute__((ext_vector_type(4)))  unsigned int u32x4;
typedef __attribute__((ext_vector_type(4)))  float  f32x4;

union Frag { v16bf v; u32x4 q[2]; };

__device__ __forceinline__ v8f wmma_bf16(const Frag& a, const Frag& b, v8f c) {
  return __builtin_amdgcn_wmma_f32_16x16x32_bf16(
      /*neg_a=*/false, a.v, /*neg_b=*/false, b.v,
      /*c_mod=*/(short)0, c, /*reuse_a=*/false, /*reuse_b=*/false);
}

// ===========================================================================
// Generic fp32-in GEMM: C[M,N] = A[M,K] @ B[K,N]  (bf16 compute, f32 accum)
// Double-buffered LDS (ping-pong), one barrier per K-step: global loads for
// tile i+1 are issued before the WMMA block of tile i.
//   mode 0 : f32  out[M,N] = acc + bias[n]               (final projection)
//   mode 1 : bf16 out as [b*16+h][row%2048][64]          (Q/K staging)
//   mode 2 : bf16 out as [b*16+h][d][2048] (transposed)  (V staging)
// Block: 128x128 tile, 256 threads = 8 waves (4 M-waves x 2 N-waves),
// each wave computes 32x64 via 2x4 WMMA accumulators. K step = 32.
// ===========================================================================
#define GBM 128
#define GBN 128
#define GBK 32
#define LDSA 40   // bf16 elems per A-tile row  (mult of 8 -> 16B aligned frags)
#define LDSB 40   // bf16 elems per B-tile row  (B staged N-major)

__global__ __launch_bounds__(256)
void gemm_bf16_wmma(const float* __restrict__ A, const float* __restrict__ B,
                    int M, int K, int N,
                    bf16* __restrict__ outBf, float* __restrict__ outF,
                    const float* __restrict__ bias, int mode)
{
  __shared__ __align__(16) bf16 lsA[2][GBM * LDSA];   // [row][k] 128x32 (+pad)
  __shared__ __align__(16) bf16 lsB[2][GBN * LDSB];   // [n][k]   128x32 (+pad)

  const int tid  = threadIdx.x;
  const int lane = tid & 31;
  const int wave = tid >> 5;
  const int wm   = wave >> 1;       // 0..3 : 32-row slab
  const int wn   = wave & 1;        // 0..1 : 64-col slab
  const int m0   = blockIdx.y * GBM;
  const int n0   = blockIdx.x * GBN;
  const int lh   = lane >> 4;       // 16-lane half select
  const int ll   = lane & 15;

  // per-thread staging coordinates (same for every K-step)
  const int arow = tid >> 3, ac4 = (tid & 7) << 2;     // A: 8 float4 per row
  const int bkk  = tid >> 5, bn4 = (tid & 31) << 2;    // B: 32 float4 per k-row

  v8f acc[2][4];
  const v8f vz = {0.f,0.f,0.f,0.f,0.f,0.f,0.f,0.f};
  #pragma unroll
  for (int i = 0; i < 2; ++i)
    #pragma unroll
    for (int j = 0; j < 4; ++j) acc[i][j] = vz;

  // ---- prologue: stage tile 0 into buffer 0 ----
  {
    #pragma unroll
    for (int i = 0; i < 4; ++i) {
      f32x4 f = *(const f32x4*)(A + (size_t)(m0 + arow + i * 32) * K + ac4);
      bf16* p = &lsA[0][(arow + i * 32) * LDSA + ac4];
      p[0] = (bf16)f.x; p[1] = (bf16)f.y; p[2] = (bf16)f.z; p[3] = (bf16)f.w;
    }
    #pragma unroll
    for (int i = 0; i < 4; ++i) {
      f32x4 f = *(const f32x4*)(B + (size_t)(bkk + i * 8) * N + n0 + bn4);
      lsB[0][(bn4 + 0) * LDSB + bkk + i * 8] = (bf16)f.x;
      lsB[0][(bn4 + 1) * LDSB + bkk + i * 8] = (bf16)f.y;
      lsB[0][(bn4 + 2) * LDSB + bkk + i * 8] = (bf16)f.z;
      lsB[0][(bn4 + 3) * LDSB + bkk + i * 8] = (bf16)f.w;
    }
  }
  __syncthreads();

  int cur = 0;
  for (int k0 = 0; k0 < K; k0 += GBK) {
    const bool hasNext = (k0 + GBK) < K;
    // ---- issue global loads for next tile early (hide behind WMMA) ----
    f32x4 ra[4], rb[4];
    if (hasNext) {
      int kn = k0 + GBK;
      #pragma unroll
      for (int i = 0; i < 4; ++i)
        ra[i] = *(const f32x4*)(A + (size_t)(m0 + arow + i * 32) * K + kn + ac4);
      #pragma unroll
      for (int i = 0; i < 4; ++i)
        rb[i] = *(const f32x4*)(B + (size_t)(kn + bkk + i * 8) * N + n0 + bn4);
      if (k0 + 2 * GBK < K)   // speculative prefetch two tiles ahead
        __builtin_prefetch(A + (size_t)(m0 + arow) * K + k0 + 2 * GBK, 0, 1);
    }

    // ---- fragments: A lane=row&15, K chunks at lh*8 and lh*8+16;
    //      B lane=col, K chunks at lh*8 and lh*8+16 ----
    Frag a[2], b[4];
    #pragma unroll
    for (int mt = 0; mt < 2; ++mt) {
      int row = wm * 32 + mt * 16 + ll;
      a[mt].q[0] = *(const u32x4*)&lsA[cur][row * LDSA + lh * 8];
      a[mt].q[1] = *(const u32x4*)&lsA[cur][row * LDSA + lh * 8 + 16];
    }
    #pragma unroll
    for (int nt = 0; nt < 4; ++nt) {
      int col = wn * 64 + nt * 16 + ll;
      b[nt].q[0] = *(const u32x4*)&lsB[cur][col * LDSB + lh * 8];
      b[nt].q[1] = *(const u32x4*)&lsB[cur][col * LDSB + lh * 8 + 16];
    }
    #pragma unroll
    for (int mt = 0; mt < 2; ++mt)
      #pragma unroll
      for (int nt = 0; nt < 4; ++nt)
        acc[mt][nt] = wmma_bf16(a[mt], b[nt], acc[mt][nt]);

    // ---- drain staged registers into the alternate buffer ----
    if (hasNext) {
      int nxt = cur ^ 1;
      #pragma unroll
      for (int i = 0; i < 4; ++i) {
        bf16* p = &lsA[nxt][(arow + i * 32) * LDSA + ac4];
        p[0] = (bf16)ra[i].x; p[1] = (bf16)ra[i].y;
        p[2] = (bf16)ra[i].z; p[3] = (bf16)ra[i].w;
      }
      #pragma unroll
      for (int i = 0; i < 4; ++i) {
        lsB[nxt][(bn4 + 0) * LDSB + bkk + i * 8] = (bf16)rb[i].x;
        lsB[nxt][(bn4 + 1) * LDSB + bkk + i * 8] = (bf16)rb[i].y;
        lsB[nxt][(bn4 + 2) * LDSB + bkk + i * 8] = (bf16)rb[i].z;
        lsB[nxt][(bn4 + 3) * LDSB + bkk + i * 8] = (bf16)rb[i].w;
      }
    }
    __syncthreads();
    cur ^= 1;
  }

  // ---- store: C/D layout  m = r + 8*lh, n = ll ----
  #pragma unroll
  for (int mt = 0; mt < 2; ++mt) {
    #pragma unroll
    for (int nt = 0; nt < 4; ++nt) {
      #pragma unroll
      for (int r = 0; r < 8; ++r) {
        float v  = acc[mt][nt][r];
        int   gm = m0 + wm * 32 + mt * 16 + r + lh * 8;
        int   gn = n0 + wn * 64 + nt * 16 + ll;
        if (mode == 0) {
          outF[(size_t)gm * N + gn] = v + (bias ? bias[gn] : 0.f);
        } else {
          int bb = gm >> 11, rr = gm & 2047;      // batch rows = 2048
          int h  = gn >> 6,  d  = gn & 63;
          if (mode == 1)
            outBf[(((size_t)(bb * 16 + h) * 2048) + rr) * 64 + d] = (bf16)v;
          else // mode 2: V transposed [bh][d][2048]
            outBf[(((size_t)(bb * 16 + h) * 64) + d) * 2048 + rr] = (bf16)v;
        }
      }
    }
  }
}

// ===========================================================================
// Flash attention, one (b,h) per blockIdx.y, 128 query rows per block,
// 4 independent waves x 32 rows, key tiles of 32, head dim 64.
// Q,K bf16 [bh][n][64]; V bf16 TRANSPOSED [bh][d][nkv]. K/V WMMA B-fragments
// are loaded straight from global memory as aligned 16B runs, so the loop has
// NO workgroup barriers; only the per-wave P transpose uses (private) LDS,
// relying on hardware-ordered same-wave DS access.
// Output O fp32 [b][q][h*64+d] (ready as A of the final projection GEMM).
// ===========================================================================
#define AKT 32    // key tile
#define LPS 40    // per-wave P row stride (bf16): 32x32 (+pad)

__global__ __launch_bounds__(128)
void attn_flash_wmma(const bf16* __restrict__ Q, const bf16* __restrict__ Kk,
                     const bf16* __restrict__ Vt, float* __restrict__ O,
                     int nq, int nkv)
{
  __shared__ __align__(16) bf16 lsP[4 * 32 * LPS];

  const int tid  = threadIdx.x;
  const int lane = tid & 31;
  const int wave = tid >> 5;
  const int lh   = lane >> 4, ll = lane & 15;
  const int bh   = blockIdx.y;
  const int q0   = blockIdx.x * 128;

  const bf16* Qb = Q  + (size_t)bh * nq * 64;
  const bf16* Kb = Kk + (size_t)bh * nkv * 64;
  const bf16* Vb = Vt + (size_t)bh * 64 * nkv;

  // Q fragments live in registers for the whole kernel (2 M-tiles x 2 K-steps)
  Frag qf[2][2];
  #pragma unroll
  for (int mt = 0; mt < 2; ++mt)
    #pragma unroll
    for (int kf = 0; kf < 2; ++kf) {
      int row = q0 + wave * 32 + mt * 16 + ll;
      int kb  = kf * 32 + lh * 8;
      qf[mt][kf].q[0] = *(const u32x4*)(Qb + (size_t)row * 64 + kb);
      qf[mt][kf].q[1] = *(const u32x4*)(Qb + (size_t)row * 64 + kb + 16);
    }

  const v8f vz = {0.f,0.f,0.f,0.f,0.f,0.f,0.f,0.f};
  v8f oacc[2][4];
  float mrun[2][8], lrun[2][8];
  #pragma unroll
  for (int mt = 0; mt < 2; ++mt) {
    #pragma unroll
    for (int nt = 0; nt < 4; ++nt) oacc[mt][nt] = vz;
    #pragma unroll
    for (int r = 0; r < 8; ++r) { mrun[mt][r] = -1e30f; lrun[mt][r] = 0.f; }
  }

  const float scale = 0.125f;   // 64^-0.5
  bf16* lsPw = &lsP[wave * 32 * LPS];

  for (int j0 = 0; j0 < nkv; j0 += AKT) {
    // ---- S = Q @ K^T : B-frags straight from global K (row key, 16B d-runs)
    v8f s[2][2] = { { vz, vz }, { vz, vz } };
    #pragma unroll
    for (int nt = 0; nt < 2; ++nt) {
      const bf16* kp = Kb + (size_t)(j0 + nt * 16 + ll) * 64 + lh * 8;
      Frag b0, b1;
      b0.q[0] = *(const u32x4*)(kp);
      b0.q[1] = *(const u32x4*)(kp + 16);
      b1.q[0] = *(const u32x4*)(kp + 32);
      b1.q[1] = *(const u32x4*)(kp + 48);
      #pragma unroll
      for (int mt = 0; mt < 2; ++mt) {
        s[mt][nt] = wmma_bf16(qf[mt][0], b0, s[mt][nt]);
        s[mt][nt] = wmma_bf16(qf[mt][1], b1, s[mt][nt]);
      }
    }

    // ---- online softmax: row m = r + 8*lh lives across 16 lanes ----
    #pragma unroll
    for (int mt = 0; mt < 2; ++mt) {
      #pragma unroll
      for (int r = 0; r < 8; ++r) {
        float s0 = s[mt][0][r] * scale;
        float s1 = s[mt][1][r] * scale;
        float mx = fmaxf(s0, s1);
        #pragma unroll
        for (int off = 1; off < 16; off <<= 1)
          mx = fmaxf(mx, __shfl_xor(mx, off, 32));
        float mnew  = fmaxf(mrun[mt][r], mx);
        float alpha = __expf(mrun[mt][r] - mnew);
        float p0    = __expf(s0 - mnew);
        float p1    = __expf(s1 - mnew);
        float rs    = p0 + p1;
        #pragma unroll
        for (int off = 1; off < 16; off <<= 1)
          rs += __shfl_xor(rs, off, 32);
        lrun[mt][r] = lrun[mt][r] * alpha + rs;
        mrun[mt][r] = mnew;
        #pragma unroll
        for (int nt = 0; nt < 4; ++nt) oacc[mt][nt][r] *= alpha;
        int m = mt * 16 + r + lh * 8;               // C-layout -> LDS (P bf16)
        lsPw[m * LPS + ll]      = (bf16)p0;
        lsPw[m * LPS + 16 + ll] = (bf16)p1;
      }
    }
    // same-wave LDS store->load kept in order by hardware (DS in-order)

    // ---- O += P @ V : P A-frags from private LDS, V B-frags from global V^T
    Frag pf[2];
    #pragma unroll
    for (int mt = 0; mt < 2; ++mt) {
      int row = mt * 16 + ll;
      pf[mt].q[0] = *(const u32x4*)&lsPw[row * LPS + lh * 8];
      pf[mt].q[1] = *(const u32x4*)&lsPw[row * LPS + lh * 8 + 16];
    }
    #pragma unroll
    for (int nt = 0; nt < 4; ++nt) {
      const bf16* vp = Vb + (size_t)(nt * 16 + ll) * nkv + j0 + lh * 8;
      Frag vf;
      vf.q[0] = *(const u32x4*)(vp);
      vf.q[1] = *(const u32x4*)(vp + 16);
      #pragma unroll
      for (int mt = 0; mt < 2; ++mt)
        oacc[mt][nt] = wmma_bf16(pf[mt], vf, oacc[mt][nt]);
    }
  }

  // ---- epilogue: O /= l, store fp32 as [b][q][h*64+d] ----
  const int b = bh >> 4, h = bh & 15;
  #pragma unroll
  for (int mt = 0; mt < 2; ++mt) {
    #pragma unroll
    for (int r = 0; r < 8; ++r) {
      float inv = 1.0f / lrun[mt][r];
      int q = q0 + wave * 32 + mt * 16 + r + lh * 8;
      #pragma unroll
      for (int nt = 0; nt < 4; ++nt) {
        int d = h * 64 + nt * 16 + ll;
        O[((size_t)b * nq + q) * 1024 + d] = oacc[mt][nt][r] * inv;
      }
    }
  }
}

// ===========================================================================
// Launch: 3 projection GEMMs -> flash attention -> output projection (+bias)
// ===========================================================================
extern "C" void kernel_launch(void* const* d_in, const int* in_sizes, int n_in,
                              void* d_out, int out_size, void* d_ws, size_t ws_size,
                              hipStream_t stream) {
  const float* x   = (const float*)d_in[0];   // [4,2048,1024]
  const float* ctx = (const float*)d_in[1];   // [4,2048,768]
  const float* Wq  = (const float*)d_in[2];   // [1024,1024]
  const float* Wk  = (const float*)d_in[3];   // [768,1024]
  const float* Wv  = (const float*)d_in[4];   // [768,1024]
  const float* Wo  = (const float*)d_in[5];   // [1024,1024]
  const float* bo  = (const float*)d_in[6];   // [1024]

  const size_t QKV = (size_t)4 * 16 * 2048 * 64;   // 8Mi elems per tensor
  bf16*  qws = (bf16*)d_ws;
  bf16*  kws = qws + QKV;
  bf16*  vws = kws + QKV;                          // V stored transposed
  float* ows = (float*)(vws + QKV);                // fp32 [8192,1024]

  dim3 gProj(1024 / GBN, 8192 / GBM);              // (8, 64)
  gemm_bf16_wmma<<<gProj, 256, 0, stream>>>(x,   Wq, 8192, 1024, 1024, qws, nullptr, nullptr, 1);
  gemm_bf16_wmma<<<gProj, 256, 0, stream>>>(ctx, Wk, 8192,  768, 1024, kws, nullptr, nullptr, 1);
  gemm_bf16_wmma<<<gProj, 256, 0, stream>>>(ctx, Wv, 8192,  768, 1024, vws, nullptr, nullptr, 2);

  attn_flash_wmma<<<dim3(2048 / 128, 64), 128, 0, stream>>>(qws, kws, vws, ows, 2048, 2048);

  gemm_bf16_wmma<<<gProj, 256, 0, stream>>>(ows, Wo, 8192, 1024, 1024, nullptr, (float*)d_out, bo, 0);
}
